// AggEdgeRemovalNet_25056839205594
// MI455X (gfx1250) — matrix operand order repaired and checked by reading, not compiled, not run on hardware
//
#include <hip/hip_runtime.h>
#include <hip/hip_bf16.h>

// ---------------------------------------------------------------------------
// Types for CDNA5 WMMA
// ---------------------------------------------------------------------------
typedef __attribute__((ext_vector_type(16))) __bf16        v16bf;
typedef __attribute__((ext_vector_type(8)))  float         v8f;
typedef __attribute__((ext_vector_type(4)))  unsigned int  v4u;

union BfFrag {
    v16bf v;
    v4u   q[2];
    unsigned short u[16];
};

__device__ __forceinline__ unsigned short f2bf(float f) {
    union { float f; unsigned int u; } c;
    c.f = f;
    unsigned int x = c.u;
    x += 0x7FFFu + ((x >> 16) & 1u);   // round-to-nearest-even
    return (unsigned short)(x >> 16);
}

#define DIMF 64

// ---------------------------------------------------------------------------
// Utility kernels
// ---------------------------------------------------------------------------
__global__ void zero_f32(float* __restrict__ p, long n) {
    long i = (long)blockIdx.x * blockDim.x + threadIdx.x;
    if (i < n) p[i] = 0.0f;
}

// Convert f32 weights -> bf16, swizzled to WMMA B-fragment-major order.
// Within one layer block of (Krows x 64):
//   k = row, col; kc = k/32; half = (k%32)/16; j = k%16; nt = col/16; L = half*16 + col%16
//   dst = (kc*128 + nt*32 + L)*16 + j
__global__ void cvt_swz(const float* __restrict__ src, unsigned short* __restrict__ dst,
                        int layers, int Krows) {
    int total = layers * Krows * 64;
    int i = blockIdx.x * blockDim.x + threadIdx.x;
    if (i >= total) return;
    int layer = i / (Krows * 64);
    int rem   = i - layer * (Krows * 64);
    int k     = rem >> 6;
    int col   = rem & 63;
    int kc    = k >> 5;
    int kr    = k & 31;
    int half  = kr >> 4;
    int j     = kr & 15;
    int nt    = col >> 4;
    int L     = half * 16 + (col & 15);
    int d     = layer * (Krows * 64) + (kc * 128 + nt * 32 + L) * 16 + j;
    dst[d] = f2bf(src[i]);
}

// h[i][j] = relu(x[i] * Wn[j] + bn[j])
__global__ void node_init(const float* __restrict__ x, const float* __restrict__ Wn,
                          const float* __restrict__ bn, float* __restrict__ h, int n) {
    long t = (long)blockIdx.x * blockDim.x + threadIdx.x;
    if (t >= (long)n * DIMF) return;
    int i = (int)(t >> 6), j = (int)(t & 63);
    float v = x[i] * Wn[j] + bn[j];
    h[t] = v > 0.f ? v : 0.f;
}

// e[ei][j] = relu(ev[ei] * We[j] + be[j])           (in_edge = 1)
__global__ void edge_init1(const float* __restrict__ ev, const float* __restrict__ We,
                           const float* __restrict__ be, float* __restrict__ e, int E_) {
    long t = (long)blockIdx.x * blockDim.x + threadIdx.x;
    if (t >= (long)E_ * DIMF) return;
    int ei = (int)(t >> 6), j = (int)(t & 63);
    float v = ev[ei] * We[j] + be[j];
    e[t] = v > 0.f ? v : 0.f;
}

// e[ei][j] = relu(ev*We[0][j] + same*We[1][j] + be[j])   (in_edge = 2)
__global__ void edge_init2(const float* __restrict__ ev, const float* __restrict__ same,
                           const float* __restrict__ We, const float* __restrict__ be,
                           float* __restrict__ e, int E_) {
    long t = (long)blockIdx.x * blockDim.x + threadIdx.x;
    if (t >= (long)E_ * DIMF) return;
    int ei = (int)(t >> 6), j = (int)(t & 63);
    float v = ev[ei] * We[j] + same[ei] * We[DIMF + j] + be[j];
    e[t] = v > 0.f ? v : 0.f;
}

// out[e] = dot(efeat[e], Wout) + bout
__global__ void edge_out(const float* __restrict__ efeat, const float* __restrict__ Wout,
                         const float* __restrict__ bout, float* __restrict__ out, int E_) {
    int e = blockIdx.x * blockDim.x + threadIdx.x;
    if (e >= E_) return;
    const float* row = efeat + (long)e * DIMF;
    float acc = 0.f;
#pragma unroll
    for (int j = 0; j < DIMF; ++j) acc += row[j] * Wout[j];
    out[e] = acc + bout[0];
}

// agg[dst[e]][j] += msg[e][j]
__global__ void scatter_agg(const float* __restrict__ msg, const int* __restrict__ dst,
                            float* __restrict__ agg, int E_) {
    long t = (long)blockIdx.x * blockDim.x + threadIdx.x;
    if (t >= (long)E_ * DIMF) return;
    int ei = (int)(t >> 6), j = (int)(t & 63);
    atomicAdd(&agg[(long)dst[ei] * DIMF + j], msg[t]);
}

// ---------------------------------------------------------------------------
// WMMA message kernel: msg = relu([h[src] | h[dst] | e] @ Wm + bm)
//   A: 16 edges x 192 (bf16, LDS)   B: 192 x 64 weights (bf16, swizzled, LDS)
//   4 waves / block, 1 wave = 16 edges, full N=64 via 4 accumulator tiles.
// ---------------------------------------------------------------------------
__global__ __launch_bounds__(128)
void msg_wmma(const float* __restrict__ h, const float* __restrict__ efeat,
              const unsigned short* __restrict__ Wbf, const float* __restrict__ bias,
              float* __restrict__ out, const int* __restrict__ src,
              const int* __restrict__ dst, int E_) {
    __shared__ v4u sWq[192 * 64 / 8];      // 24 KB, swizzled fragment-major
    __shared__ v4u sAq[4][16 * 192 / 8];   // 4 x 6 KB
    unsigned short* sW = (unsigned short*)sWq;
    const int tid  = threadIdx.x;
    const int wave = tid >> 5;
    const int lane = tid & 31;

    __builtin_prefetch(Wbf, 0, 0);         // global_prefetch_b8

    // weights are pre-swizzled in global: straight 128-bit copy
    {
        const v4u* gw = (const v4u*)Wbf;
        for (int i = tid; i < 192 * 64 / 8; i += 128) sWq[i] = gw[i];
    }

    unsigned short* sA = (unsigned short*)sAq[wave];
    const int ebase = (blockIdx.x * 4 + wave) * 16;
    for (int idx = lane; idx < 16 * 192; idx += 32) {
        int r  = idx / 192;
        int kk = idx - r * 192;
        int eidx = ebase + r;
        float v = 0.f;
        if (eidx < E_) {
            if (kk < 64)       v = h[(long)src[eidx] * DIMF + kk];
            else if (kk < 128) v = h[(long)dst[eidx] * DIMF + (kk - 64)];
            else               v = efeat[(long)eidx * DIMF + (kk - 128)];
        }
        sA[idx] = f2bf(v);
    }
    __syncthreads();

    const int row  = lane & 15;
    const int half = lane >> 4;

    v8f acc0 = {0.f,0.f,0.f,0.f,0.f,0.f,0.f,0.f};
    v8f acc[4] = {acc0, acc0, acc0, acc0};

    for (int kc = 0; kc < 6; ++kc) {
        BfFrag a;
        // elements 0..7  <-> k = kc*32 + half*16 + e      (16B contiguous)
        // elements 8..15 <-> k = kc*32 + half*16 + 8 + e  (16B contiguous)
        a.q[0] = *(const v4u*)&sA[row * 192 + kc * 32 + half * 16];
        a.q[1] = *(const v4u*)&sA[row * 192 + kc * 32 + half * 16 + 8];
#pragma unroll
        for (int nt = 0; nt < 4; ++nt) {
            BfFrag b;
            const v4u* bp = (const v4u*)&sW[(kc * 128 + nt * 32 + lane) * 16];
            b.q[0] = bp[0];
            b.q[1] = bp[1];
            acc[nt] = __builtin_amdgcn_wmma_f32_16x16x32_bf16(
                false, a.v, false, b.v, (short)0, acc[nt], false, false);
        }
    }

#pragma unroll
    for (int nt = 0; nt < 4; ++nt) {
        int col = nt * 16 + row;
        float bv = bias[col];
#pragma unroll
        for (int r = 0; r < 8; ++r) {
            int m = r + half * 8;
            int eidx = ebase + m;
            if (eidx < E_) {
                float v = acc[nt][r] + bv;
                out[(long)eidx * DIMF + col] = v > 0.f ? v : 0.f;
            }
        }
    }
}

// ---------------------------------------------------------------------------
// WMMA node-update kernel: h' = relu([h | agg] @ Wh + bh)   (K = 128)
// ---------------------------------------------------------------------------
__global__ __launch_bounds__(128)
void hupd_wmma(const float* __restrict__ h, const float* __restrict__ agg,
               const unsigned short* __restrict__ Wbf, const float* __restrict__ bias,
               float* __restrict__ out, int n) {
    __shared__ v4u sWq[128 * 64 / 8];      // 16 KB, swizzled
    __shared__ v4u sAq[4][16 * 128 / 8];   // 4 x 4 KB
    unsigned short* sW = (unsigned short*)sWq;
    const int tid  = threadIdx.x;
    const int wave = tid >> 5;
    const int lane = tid & 31;

    {
        const v4u* gw = (const v4u*)Wbf;
        for (int i = tid; i < 128 * 64 / 8; i += 128) sWq[i] = gw[i];
    }

    unsigned short* sA = (unsigned short*)sAq[wave];
    const int nbase = (blockIdx.x * 4 + wave) * 16;
    for (int idx = lane; idx < 16 * 128; idx += 32) {
        int r  = idx >> 7;
        int kk = idx & 127;
        int node = nbase + r;
        float v = 0.f;
        if (node < n) {
            if (kk < 64) v = h[(long)node * DIMF + kk];
            else         v = agg[(long)node * DIMF + (kk - 64)];
        }
        sA[idx] = f2bf(v);
    }
    __syncthreads();

    const int row  = lane & 15;
    const int half = lane >> 4;

    v8f acc0 = {0.f,0.f,0.f,0.f,0.f,0.f,0.f,0.f};
    v8f acc[4] = {acc0, acc0, acc0, acc0};

    for (int kc = 0; kc < 4; ++kc) {
        BfFrag a;
        a.q[0] = *(const v4u*)&sA[row * 128 + kc * 32 + half * 16];
        a.q[1] = *(const v4u*)&sA[row * 128 + kc * 32 + half * 16 + 8];
#pragma unroll
        for (int nt = 0; nt < 4; ++nt) {
            BfFrag b;
            const v4u* bp = (const v4u*)&sW[(kc * 128 + nt * 32 + lane) * 16];
            b.q[0] = bp[0];
            b.q[1] = bp[1];
            acc[nt] = __builtin_amdgcn_wmma_f32_16x16x32_bf16(
                false, a.v, false, b.v, (short)0, acc[nt], false, false);
        }
    }

#pragma unroll
    for (int nt = 0; nt < 4; ++nt) {
        int col = nt * 16 + row;
        float bv = bias[col];
#pragma unroll
        for (int r = 0; r < 8; ++r) {
            int m = r + half * 8;
            int node = nbase + m;
            if (node < n) {
                float v = acc[nt][r] + bv;
                out[(long)node * DIMF + col] = v > 0.f ? v : 0.f;
            }
        }
    }
}

// ---------------------------------------------------------------------------
// Sorting (bitonic, descending) + union-find coarsening
// ---------------------------------------------------------------------------
__global__ void sort_init(const float* __restrict__ scores, float* __restrict__ key,
                          int* __restrict__ idx, int E_, int SZ) {
    int i = blockIdx.x * blockDim.x + threadIdx.x;
    if (i >= SZ) return;
    if (i < E_) { key[i] = scores[i]; idx[i] = i; }
    else        { key[i] = -3.402823466e38f; idx[i] = 0; }
}

__global__ void bitonic_step(float* __restrict__ key, int* __restrict__ idx,
                             int j, int k, int SZ) {
    int i = blockIdx.x * blockDim.x + threadIdx.x;
    if (i >= SZ) return;
    int ixj = i ^ j;
    if (ixj > i) {
        float ki = key[i], kj = key[ixj];
        bool descBlock = ((i & k) == 0);
        bool doSwap = descBlock ? (ki < kj) : (ki > kj);
        if (doSwap) {
            key[i] = kj; key[ixj] = ki;
            int t = idx[i]; idx[i] = idx[ixj]; idx[ixj] = t;
        }
    }
}

__global__ void uf_init(int* __restrict__ parent, int* __restrict__ rnk, int n) {
    int i = blockIdx.x * blockDim.x + threadIdx.x;
    if (i < n) { parent[i] = i; rnk[i] = 0; }
}

__device__ __forceinline__ int uf_find(const int* parent, int v) {
    while (parent[v] != v) v = parent[v];
    return v;
}

// Sequential greedy contraction (matches reference while_loop semantics).
__global__ void uf_run(const int* __restrict__ order, const int* __restrict__ src,
                       const int* __restrict__ dst, int* parent, int* rnk,
                       const int* __restrict__ kptr, int n, int E_) {
    if (threadIdx.x != 0 || blockIdx.x != 0) return;
    int kk = kptr[0];
    int ns = n;
    for (int i = 0; i < E_ && ns > kk; ++i) {
        int e = order[i];
        int a = uf_find(parent, src[e]);
        int b = uf_find(parent, dst[e]);
        if (a != b) {
            int ra = rnk[a], rb = rnk[b];
            int hi = (ra >= rb) ? a : b;
            int lo = (ra >= rb) ? b : a;
            parent[lo] = hi;
            if (ra == rb) rnk[hi] = ra + 1;
            --ns;
        }
    }
}

// col_of = cumsum(parent == arange) - 1  (sequential; n is small)
__global__ void colof_seq(const int* __restrict__ parent, int* __restrict__ colof, int n) {
    if (threadIdx.x != 0 || blockIdx.x != 0) return;
    int c = 0;
    for (int v = 0; v < n; ++v) {
        if (parent[v] == v) ++c;
        colof[v] = c - 1;
    }
}

__global__ void cols_map(const int* __restrict__ parent, const int* __restrict__ colof,
                         int* __restrict__ cols, int n) {
    int v = blockIdx.x * blockDim.x + threadIdx.x;
    if (v >= n) return;
    cols[v] = colof[uf_find(parent, v)];
}

__global__ void same_kernel(const int* __restrict__ cols, const int* __restrict__ src,
                            const int* __restrict__ dst, float* __restrict__ same, int E_) {
    int e = blockIdx.x * blockDim.x + threadIdx.x;
    if (e >= E_) return;
    same[e] = (cols[src[e]] == cols[dst[e]]) ? 1.0f : 0.0f;
}

__global__ void cols_to_out(const int* __restrict__ cols, float* __restrict__ out, int n) {
    int i = blockIdx.x * blockDim.x + threadIdx.x;
    if (i < n) out[i] = (float)cols[i];
}

// P[src[e], cols[dst[e]]] += p_edges[e]   (mode='drop' on OOB col)
__global__ void scatterP(const int* __restrict__ src, const int* __restrict__ dst,
                         const int* __restrict__ cols, const float* __restrict__ pe,
                         float* __restrict__ P, int E_, int K) {
    int e = blockIdx.x * blockDim.x + threadIdx.x;
    if (e >= E_) return;
    int c = cols[dst[e]];
    if ((unsigned)c < (unsigned)K)
        atomicAdd(&P[(long)src[e] * K + c], pe[e]);
}

// ---------------------------------------------------------------------------
// Host orchestration
// ---------------------------------------------------------------------------
static inline int gdiv(long a, int b) { return (int)((a + b - 1) / b); }

extern "C" void kernel_launch(void* const* d_in, const int* in_sizes, int n_in,
                              void* d_out, int out_size, void* d_ws, size_t ws_size,
                              hipStream_t stream) {
    const int*   edge_index = (const int*)d_in[0];
    const float* edge_vals  = (const float*)d_in[1];
    const float* x          = (const float*)d_in[2];
    const int*   kptr       = (const int*)d_in[3];
    // edgenet params
    const float* Wn1 = (const float*)d_in[4];
    const float* bn1 = (const float*)d_in[5];
    const float* We1 = (const float*)d_in[6];
    const float* be1 = (const float*)d_in[7];
    const float* Wm1 = (const float*)d_in[8];
    const float* bm1 = (const float*)d_in[9];
    const float* Wh1 = (const float*)d_in[10];
    const float* bh1 = (const float*)d_in[11];
    const float* Wo1 = (const float*)d_in[12];
    const float* bo1 = (const float*)d_in[13];
    // pnet params
    const float* Wn2 = (const float*)d_in[14];
    const float* bn2 = (const float*)d_in[15];
    const float* We2 = (const float*)d_in[16];
    const float* be2 = (const float*)d_in[17];
    const float* Wm2 = (const float*)d_in[18];
    const float* bm2 = (const float*)d_in[19];
    const float* Wh2 = (const float*)d_in[20];
    const float* bh2 = (const float*)d_in[21];
    const float* Wo2 = (const float*)d_in[22];
    const float* bo2 = (const float*)d_in[23];

    const int E_ = in_sizes[0] / 2;
    const int Nn = in_sizes[2];
    const int K  = (int)(((long)out_size - Nn - E_) / Nn);   // out = cols | P | a_edges
    const int* src = edge_index;
    const int* dst = edge_index + E_;

    int SZ = 1;
    while (SZ < E_) SZ <<= 1;

    // workspace carve
    char* w = (char*)d_ws;
    size_t off = 0;
    auto carve = [&](size_t bytes) -> void* {
        void* p = w + off;
        off = (off + bytes + 255) & ~(size_t)255;
        return p;
    };
    float* hA   = (float*)carve((size_t)Nn * DIMF * 4);
    float* hB   = (float*)carve((size_t)Nn * DIMF * 4);
    float* agg  = (float*)carve((size_t)Nn * DIMF * 4);
    float* eA   = (float*)carve((size_t)E_ * DIMF * 4);
    float* eB   = (float*)carve((size_t)E_ * DIMF * 4);
    float* ped  = (float*)carve((size_t)E_ * 4);
    float* key  = (float*)carve((size_t)SZ * 4);
    int*   idxb = (int*)  carve((size_t)SZ * 4);
    int*   par  = (int*)  carve((size_t)Nn * 4);
    int*   rnk  = (int*)  carve((size_t)Nn * 4);
    int*   colf = (int*)  carve((size_t)Nn * 4);
    int*   cols = (int*)  carve((size_t)Nn * 4);
    float* same = (float*)carve((size_t)E_ * 4);
    unsigned short* Wm1b = (unsigned short*)carve((size_t)4 * 192 * 64 * 2);
    unsigned short* Wh1b = (unsigned short*)carve((size_t)4 * 128 * 64 * 2);
    unsigned short* Wm2b = (unsigned short*)carve((size_t)4 * 192 * 64 * 2);
    unsigned short* Wh2b = (unsigned short*)carve((size_t)4 * 128 * 64 * 2);
    (void)ws_size; (void)n_in;

    float* outF = (float*)d_out;
    float* Pout = outF + Nn;                       // N x K
    float* aed  = outF + Nn + (size_t)Nn * K;      // E a_edges

    // zero output (cols | P | a_edges)
    zero_f32<<<gdiv((long)out_size, 256), 256, 0, stream>>>(outF, (long)out_size);

    // weights -> bf16, swizzled to fragment order
    cvt_swz<<<gdiv(4 * 192 * 64, 256), 256, 0, stream>>>(Wm1, Wm1b, 4, 192);
    cvt_swz<<<gdiv(4 * 128 * 64, 256), 256, 0, stream>>>(Wh1, Wh1b, 4, 128);
    cvt_swz<<<gdiv(4 * 192 * 64, 256), 256, 0, stream>>>(Wm2, Wm2b, 4, 192);
    cvt_swz<<<gdiv(4 * 128 * 64, 256), 256, 0, stream>>>(Wh2, Wh2b, 4, 128);

    const int msgBlocks  = gdiv((long)E_, 64);
    const int hBlocks    = gdiv((long)Nn, 64);
    const long nd = (long)Nn * DIMF, ed = (long)E_ * DIMF;

    // ---------------- edgenet MPNN ----------------
    node_init <<<gdiv(nd, 256), 256, 0, stream>>>(x, Wn1, bn1, hA, Nn);
    edge_init1<<<gdiv(ed, 256), 256, 0, stream>>>(edge_vals, We1, be1, eA, E_);
    {
        float* hc = hA; float* hn = hB; float* ec = eA; float* en = eB;
        for (int l = 0; l < 4; ++l) {
            msg_wmma<<<msgBlocks, 128, 0, stream>>>(hc, ec, Wm1b + (size_t)l * 192 * 64,
                                                    bm1 + l * 64, en, src, dst, E_);
            zero_f32<<<gdiv(nd, 256), 256, 0, stream>>>(agg, nd);
            scatter_agg<<<gdiv(ed, 256), 256, 0, stream>>>(en, dst, agg, E_);
            hupd_wmma<<<hBlocks, 128, 0, stream>>>(hc, agg, Wh1b + (size_t)l * 128 * 64,
                                                   bh1 + l * 64, hn, Nn);
            float* t;
            t = hc; hc = hn; hn = t;
            t = ec; ec = en; en = t;
        }
        edge_out<<<gdiv((long)E_, 256), 256, 0, stream>>>(ec, Wo1, bo1, aed, E_);
    }

    // ---------------- coarsen: sort (desc) + union-find ----------------
    sort_init<<<gdiv((long)SZ, 256), 256, 0, stream>>>(aed, key, idxb, E_, SZ);
    for (int kk = 2; kk <= SZ; kk <<= 1)
        for (int j = kk >> 1; j > 0; j >>= 1)
            bitonic_step<<<gdiv((long)SZ, 256), 256, 0, stream>>>(key, idxb, j, kk, SZ);
    uf_init<<<gdiv((long)Nn, 256), 256, 0, stream>>>(par, rnk, Nn);
    uf_run<<<1, 1, 0, stream>>>(idxb, src, dst, par, rnk, kptr, Nn, E_);
    colof_seq<<<1, 1, 0, stream>>>(par, colf, Nn);
    cols_map<<<gdiv((long)Nn, 256), 256, 0, stream>>>(par, colf, cols, Nn);
    cols_to_out<<<gdiv((long)Nn, 256), 256, 0, stream>>>(cols, outF, Nn);
    same_kernel<<<gdiv((long)E_, 256), 256, 0, stream>>>(cols, src, dst, same, E_);

    // ---------------- pnet MPNN ----------------
    node_init <<<gdiv(nd, 256), 256, 0, stream>>>(x, Wn2, bn2, hA, Nn);
    edge_init2<<<gdiv(ed, 256), 256, 0, stream>>>(edge_vals, same, We2, be2, eA, E_);
    {
        float* hc = hA; float* hn = hB; float* ec = eA; float* en = eB;
        for (int l = 0; l < 4; ++l) {
            msg_wmma<<<msgBlocks, 128, 0, stream>>>(hc, ec, Wm2b + (size_t)l * 192 * 64,
                                                    bm2 + l * 64, en, src, dst, E_);
            zero_f32<<<gdiv(nd, 256), 256, 0, stream>>>(agg, nd);
            scatter_agg<<<gdiv(ed, 256), 256, 0, stream>>>(en, dst, agg, E_);
            hupd_wmma<<<hBlocks, 128, 0, stream>>>(hc, agg, Wh2b + (size_t)l * 128 * 64,
                                                   bh2 + l * 64, hn, Nn);
            float* t;
            t = hc; hc = hn; hn = t;
            t = ec; ec = en; en = t;
        }
        edge_out<<<gdiv((long)E_, 256), 256, 0, stream>>>(ec, Wo2, bo2, ped, E_);
    }

    // ---------------- P scatter ----------------
    scatterP<<<gdiv((long)E_, 256), 256, 0, stream>>>(src, dst, cols, ped, Pout, E_, K);
}
